// GGAT_RES_16363825398383
// MI455X (gfx1250) — matrix operand order repaired and checked by reading, not compiled, not run on hardware
//
#include <hip/hip_runtime.h>

// ---------------------------------------------------------------------------
// Gated multi-head GAT (4 layers + GRU gating + sigmoid head) for MI455X.
// Dense GEMMs use v_wmma_f32_16x16x32_bf16 (bf16 in, f32 accumulate);
// each wave computes a 16x64 C strip (4 accumulators share one A fragment),
// K=128 fully unrolled -> 16 WMMAs back-to-back per wave. Weights are
// repacked once per layer into bf16 [Ncols x K] so B fragments are single
// aligned 32B contiguous loads (no converts, no strided traffic in-loop).
// Graph ops (edge softmax / scatter aggregation) are bandwidth kernels sized
// so Wh+agg (~82MB) stay L2-resident (192MB on MI455X).
// ---------------------------------------------------------------------------

#define D_DIM 128
#define NEG_SLOPE 0.2f

typedef __attribute__((ext_vector_type(16))) __bf16 v16bf;
typedef __attribute__((ext_vector_type(16))) float  v16f;
typedef __attribute__((ext_vector_type(8)))  float  v8f;

// native packed f32->bf16 converts (RNE) via vector fptrunc
__device__ __forceinline__ v16bf pack16(const float4& x0, const float4& x1,
                                        const float4& x2, const float4& x3) {
  v16f f;
  f[0]=x0.x;  f[1]=x0.y;  f[2]=x0.z;  f[3]=x0.w;
  f[4]=x1.x;  f[5]=x1.y;  f[6]=x1.z;  f[7]=x1.w;
  f[8]=x2.x;  f[9]=x2.y;  f[10]=x2.z; f[11]=x2.w;
  f[12]=x3.x; f[13]=x3.y; f[14]=x3.z; f[15]=x3.w;
  return __builtin_convertvector(f, v16bf);
}

// ---------------------------------------------------------------------------
// Weight repack kernels (run once per layer / once per launch; tiny).
//   conv_tr_k : Wt[n*K + k] = bf16( W[k*Ncols + n] )   (transpose KxN -> NxK)
//   conv_k    : Y[i]        = bf16( X[i] )             (layout already NxK)
// ---------------------------------------------------------------------------
__global__ void conv_tr_k(const float* __restrict__ W, __bf16* __restrict__ Wt,
                          int K, int Ncols)
{
  int idx = blockIdx.x * blockDim.x + threadIdx.x;   // idx = n*K + k
  if (idx >= K * Ncols) return;
  int n = idx / K, k = idx - n * K;
  Wt[idx] = (__bf16)W[(size_t)k * Ncols + n];
}

__global__ void conv_k(const float* __restrict__ X, __bf16* __restrict__ Y, int total)
{
  int idx = blockIdx.x * blockDim.x + threadIdx.x;
  if (idx >= total) return;
  Y[idx] = (__bf16)X[idx];
}

// ---------------------------------------------------------------------------
// GEMM:  C[m,n] = sum_k A[m,k] * Bt[n,k]  (+ bias[n])
// A: f32 [M x K] row-major (converted to bf16 in-register).
// Bt: bf16 [Ncols x K] row-major (pre-packed weights).
// One wave computes a 16x64 C strip; K templated (128) -> fully unrolled.
// Requires M%16==0, Ncols%64==0, K%32==0 (N=20000=1250*16, D=128, 3D=384).
// ---------------------------------------------------------------------------
template<int K>
__global__ void __launch_bounds__(128)
wmma_gemm_bf16(const float* __restrict__ A, int lda,
               const __bf16* __restrict__ Bt,
               float* __restrict__ C, int ldc,
               const float* __restrict__ bias,
               int M, int Ncols)
{
  const int lane = threadIdx.x & 31;
  const int wave = threadIdx.x >> 5;
  const int n64    = Ncols >> 6;
  const int mtiles = M >> 4;
  const int tile = blockIdx.x * 4 + wave;          // uniform per wave: EXEC stays full
  if (tile >= mtiles * n64) return;
  const int mt = tile / n64;
  const int ng = tile % n64;

  const int hl  = lane >> 4;     // lane half (0/1)
  const int l16 = lane & 15;
  const int arow = mt * 16 + l16;
  const int bcol0 = ng * 64 + l16;                 // column for B-frag j: bcol0 + 16*j

  v8f acc[4] = {v8f{}, v8f{}, v8f{}, v8f{}};
#pragma unroll
  for (int k0 = 0; k0 < K; k0 += 32) {
    // ---- A tile 16x32 bf16 (ISA 7.12.2): lane half picks K blocks ----
    const int kbase = k0 + hl * 16;                // 32B aligned (lda mult of 8)
    const float* ap = A + (size_t)arow * lda + kbase;
    v16bf a = pack16(*reinterpret_cast<const float4*>(ap),
                     *reinterpret_cast<const float4*>(ap + 4),
                     *reinterpret_cast<const float4*>(ap + 8),
                     *reinterpret_cast<const float4*>(ap + 12));

    // ---- B tiles 32x16 bf16: lanes 0-15 K=k0..k0+15, lanes 16-31 K=+16 ----
    const int kb0 = k0 + hl * 16;
#pragma unroll
    for (int j = 0; j < 4; ++j) {
      const int bcol = bcol0 + 16 * j;
      // 16 contiguous bf16 = 32B, 32B-aligned (K mult of 16, bcol*K mult of 16)
      v16bf b = *reinterpret_cast<const v16bf*>(Bt + (size_t)bcol * K + kb0);
      acc[j] = __builtin_amdgcn_wmma_f32_16x16x32_bf16(false, a, false, b,
                                                       (short)0, acc[j], false, false);
    }
  }

#pragma unroll
  for (int j = 0; j < 4; ++j) {
    const int bcol = bcol0 + 16 * j;
    const float bv = bias ? bias[bcol] : 0.0f;
#pragma unroll
    for (int r = 0; r < 8; ++r) {                  // C: VGPR r -> row r (+8 for hi half)
      const int m = mt * 16 + r + hl * 8;
      C[(size_t)m * ldc + bcol] = acc[j][r] + bv;
    }
  }
}

// ---------------------------------------------------------------------------
// es[n,h] = <Wh[n,h,:], a_src[h,:]> ; ed likewise
// ---------------------------------------------------------------------------
__global__ void attn_scores_k(const float* __restrict__ Wh,
                              const float* __restrict__ as_,
                              const float* __restrict__ ad_,
                              float* __restrict__ es, float* __restrict__ edv,
                              int total, int Hl)
{
  int idx = blockIdx.x * blockDim.x + threadIdx.x;   // idx = n*Hl + h
  if (idx >= total) return;
  int h = idx % Hl;
  const float* w = Wh + (size_t)idx * D_DIM;
  const float* s = as_ + (size_t)h * D_DIM;
  const float* d = ad_ + (size_t)h * D_DIM;
  float accs = 0.f, accd = 0.f;
  for (int i = 0; i < D_DIM; ++i) { float v = w[i]; accs += v * s[i]; accd += v * d[i]; }
  es[idx] = accs; edv[idx] = accd;
}

// monotonic float <-> uint mapping for atomicMax over floats (incl. negatives)
__device__ __forceinline__ unsigned fmapU(float f) {
  unsigned u = __float_as_uint(f);
  return (u & 0x80000000u) ? ~u : (u | 0x80000000u);
}
__device__ __forceinline__ float funmapU(unsigned u) {
  return (u & 0x80000000u) ? __uint_as_float(u & 0x7FFFFFFFu) : __uint_as_float(~u);
}

__global__ void edge_max_k(const float* __restrict__ es, const float* __restrict__ edv,
                           const int* __restrict__ src, const int* __restrict__ dst,
                           float* __restrict__ escore, unsigned* __restrict__ mInt,
                           int E, int Hl)
{
  int idx = blockIdx.x * blockDim.x + threadIdx.x;   // idx = e*Hl + h
  if (idx >= E * Hl) return;
  int e = idx / Hl, h = idx - e * Hl;
  int s = src[e], d = dst[e];
  float v = es[(size_t)s * Hl + h] + edv[(size_t)d * Hl + h];
  v = (v > 0.f) ? v : NEG_SLOPE * v;                 // leaky_relu
  escore[idx] = v;
  atomicMax(&mInt[(size_t)d * Hl + h], fmapU(v));    // segment max
}

__global__ void edge_expsum_k(const float* __restrict__ escore, float* __restrict__ wbuf,
                              const unsigned* __restrict__ mInt,
                              const int* __restrict__ dst, float* __restrict__ denom,
                              int E, int Hl)
{
  int idx = blockIdx.x * blockDim.x + threadIdx.x;
  if (idx >= E * Hl) return;
  int e = idx / Hl, h = idx - e * Hl;
  int d = dst[e];
  float m = funmapU(mInt[(size_t)d * Hl + h]);
  float w = __expf(escore[idx] - m);
  wbuf[idx] = w;
  atomicAdd(&denom[(size_t)d * Hl + h], w);          // segment sum
}

// one block per edge; lanes coalesced over D; L2-resident Wh/agg
__global__ void aggregate_k(const float* __restrict__ wbuf, const float* __restrict__ denom,
                            const float* __restrict__ Wh,
                            const int* __restrict__ src, const int* __restrict__ dst,
                            float* __restrict__ agg, int Hl)
{
  int e = blockIdx.x;
  int s = src[e], d = dst[e];
  int tot = Hl * D_DIM;
  for (int idx = threadIdx.x; idx < tot; idx += blockDim.x) {
    int h = idx / D_DIM, dd = idx - h * D_DIM;
    float alpha = wbuf[(size_t)e * Hl + h] / denom[(size_t)d * Hl + h];
    atomicAdd(&agg[((size_t)d * Hl + h) * D_DIM + dd],
              alpha * Wh[((size_t)s * Hl + h) * D_DIM + dd]);
  }
}

__global__ void headmean_elu_k(const float* __restrict__ agg, float* __restrict__ out,
                               int total, int Hl)
{
  int idx = blockIdx.x * blockDim.x + threadIdx.x;   // idx = n*D + d
  if (idx >= total) return;
  int n = idx / D_DIM, d = idx - n * D_DIM;
  float a = 0.f;
  for (int h = 0; h < Hl; ++h) a += agg[((size_t)n * Hl + h) * D_DIM + d];
  a /= (float)Hl;
  out[idx] = (a > 0.f) ? a : (expf(a) - 1.f);        // elu
}

__global__ void gru_combine_k(const float* __restrict__ gi, const float* __restrict__ gh,
                              const float* __restrict__ hprev, float* __restrict__ hnew,
                              int total)
{
  int idx = blockIdx.x * blockDim.x + threadIdx.x;
  if (idx >= total) return;
  int n = idx / D_DIM, d = idx - n * D_DIM;
  const float* giN = gi + (size_t)n * 3 * D_DIM;
  const float* ghN = gh + (size_t)n * 3 * D_DIM;
  float r  = 1.f / (1.f + expf(-(giN[d] + ghN[d])));
  float z  = 1.f / (1.f + expf(-(giN[D_DIM + d] + ghN[D_DIM + d])));
  float nn = tanhf(giN[2 * D_DIM + d] + r * ghN[2 * D_DIM + d]);
  hnew[idx] = (1.f - z) * nn + z * hprev[idx];
}

__global__ void out_head_k(const float* __restrict__ h, const float* __restrict__ W5,
                           const float* __restrict__ b5, float* __restrict__ out, int N)
{
  int n = blockIdx.x * blockDim.x + threadIdx.x;
  if (n >= N) return;
  float s = 0.f;
  for (int d = 0; d < D_DIM; ++d) s += h[(size_t)n * D_DIM + d] * W5[d];
  s += b5[0];
  out[n] = 1.f / (1.f + expf(-s));
}

// ---------------------------------------------------------------------------
struct Ws {
  float *Wh, *es, *edv, *escore, *wbuf, *denom, *agg, *gi, *gh, *outbuf;
  unsigned* mInt;
  __bf16 *Wb, *Wihb, *Whhb;
};

static void run_layer(const float* hin, float* hout,
                      const float* W, const float* as_, const float* ad_,
                      int Hl, bool gru,
                      const float* bih, const float* bhh,
                      const Ws& ws, const int* src, const int* dst,
                      int N, int E, hipStream_t stream)
{
  const int D = D_DIM;
  hipMemsetAsync(ws.mInt,  0, (size_t)N * Hl * sizeof(unsigned), stream);
  hipMemsetAsync(ws.denom, 0, (size_t)N * Hl * sizeof(float),    stream);
  hipMemsetAsync(ws.agg,   0, (size_t)N * Hl * D * sizeof(float), stream);

  // repack this layer's weights: W[h] (KxN) -> bf16 (NxK)
  for (int h = 0; h < Hl; ++h)
    conv_tr_k<<<(D * D + 255) / 256, 256, 0, stream>>>(
        W + (size_t)h * D * D, ws.Wb + (size_t)h * D * D, D, D);

  // Wh[n,h,:] = h[n,:] @ W[h,:,:]
  {
    int tiles = (N / 16) * (D / 64);
    int blocks = (tiles + 3) / 4;
    for (int h = 0; h < Hl; ++h)
      wmma_gemm_bf16<D_DIM><<<blocks, 128, 0, stream>>>(
          hin, D, ws.Wb + (size_t)h * D * D,
          ws.Wh + (size_t)h * D, Hl * D, nullptr, N, D);
  }
  attn_scores_k<<<(N * Hl + 255) / 256, 256, 0, stream>>>(ws.Wh, as_, ad_, ws.es, ws.edv, N * Hl, Hl);
  edge_max_k   <<<(E * Hl + 255) / 256, 256, 0, stream>>>(ws.es, ws.edv, src, dst, ws.escore, ws.mInt, E, Hl);
  edge_expsum_k<<<(E * Hl + 255) / 256, 256, 0, stream>>>(ws.escore, ws.wbuf, ws.mInt, dst, ws.denom, E, Hl);
  aggregate_k  <<<E, 128, 0, stream>>>(ws.wbuf, ws.denom, ws.Wh, src, dst, ws.agg, Hl);
  headmean_elu_k<<<(N * D + 255) / 256, 256, 0, stream>>>(ws.agg, gru ? ws.outbuf : hout, N * D, Hl);

  if (gru) {
    int tiles = (N / 16) * (3 * D / 64);
    int blocks = (tiles + 3) / 4;
    // gi = out @ Wih^T + bih ; gh = h @ Whh^T + bhh   (torch GRUCell layout)
    wmma_gemm_bf16<D_DIM><<<blocks, 128, 0, stream>>>(ws.outbuf, D, ws.Wihb, ws.gi, 3 * D, bih, N, 3 * D);
    wmma_gemm_bf16<D_DIM><<<blocks, 128, 0, stream>>>(hin,       D, ws.Whhb, ws.gh, 3 * D, bhh, N, 3 * D);
    gru_combine_k<<<(N * D + 255) / 256, 256, 0, stream>>>(ws.gi, ws.gh, hin, hout, N * D);
  }
}

extern "C" void kernel_launch(void* const* d_in, const int* in_sizes, int n_in,
                              void* d_out, int out_size, void* d_ws, size_t ws_size,
                              hipStream_t stream)
{
  const int D = D_DIM;
  const int N = in_sizes[0] / D;
  const int E = in_sizes[1] / 2;

  const float* h0   = (const float*)d_in[0];
  const int*   ei   = (const int*)d_in[1];
  const int*   src  = ei;
  const int*   dst  = ei + E;
  const float* Wl[4]  = {(const float*)d_in[2], (const float*)d_in[5], (const float*)d_in[8],  (const float*)d_in[11]};
  const float* asl[4] = {(const float*)d_in[3], (const float*)d_in[6], (const float*)d_in[9],  (const float*)d_in[12]};
  const float* adl[4] = {(const float*)d_in[4], (const float*)d_in[7], (const float*)d_in[10], (const float*)d_in[13]};
  int Hc[4];
  for (int l = 0; l < 4; ++l) Hc[l] = in_sizes[3 + 3 * l] / D;  // as{l} is [H,D]
  const float* Wih = (const float*)d_in[14];
  const float* Whh = (const float*)d_in[15];
  const float* bih = (const float*)d_in[16];
  const float* bhh = (const float*)d_in[17];
  const float* W5  = (const float*)d_in[18];
  const float* b5  = (const float*)d_in[19];

  // ---- scratch layout (bump allocator, 256B aligned) ----
  char* base = (char*)d_ws;
  size_t off = 0;
  auto bump = [&](size_t bytes) -> void* {
    void* p = base + off;
    off = (off + bytes + 255) & ~(size_t)255;
    return p;
  };
  const int Hmax = 4;
  Ws ws;
  ws.Wh     = (float*)bump((size_t)N * Hmax * D * 4);
  ws.es     = (float*)bump((size_t)N * Hmax * 4);
  ws.edv    = (float*)bump((size_t)N * Hmax * 4);
  ws.escore = (float*)bump((size_t)E * Hmax * 4);
  ws.wbuf   = (float*)bump((size_t)E * Hmax * 4);
  ws.mInt   = (unsigned*)bump((size_t)N * Hmax * 4);
  ws.denom  = (float*)bump((size_t)N * Hmax * 4);
  // agg (<= N*Hmax*D floats) and gi (N*3D floats) never live simultaneously
  float* overlay = (float*)bump((size_t)N * (Hmax > 3 ? Hmax : 3) * D * 4 + (size_t)N * D * 4);
  ws.agg = overlay;
  ws.gi  = overlay;
  ws.gh     = (float*)bump((size_t)N * 3 * D * 4);
  ws.outbuf = (float*)bump((size_t)N * D * 4);
  float* hA = (float*)bump((size_t)N * D * 4);
  float* hB = (float*)bump((size_t)N * D * 4);
  ws.Wb   = (__bf16*)bump((size_t)Hmax * D * D * 2);
  ws.Wihb = (__bf16*)bump((size_t)3 * D * D * 2);
  ws.Whhb = (__bf16*)bump((size_t)3 * D * D * 2);

  // pre-convert shared GRU weights (already [Ncols x K] torch layout)
  conv_k<<<(3 * D * D + 255) / 256, 256, 0, stream>>>(Wih, ws.Wihb, 3 * D * D);
  conv_k<<<(3 * D * D + 255) / 256, 256, 0, stream>>>(Whh, ws.Whhb, 3 * D * D);

  hipMemcpyAsync(hA, h0, (size_t)N * D * 4, hipMemcpyDeviceToDevice, stream);

  // layer 1: no GRU, in-place on hA (hin only read before hout is written)
  run_layer(hA, hA, Wl[0], asl[0], adl[0], Hc[0], false, bih, bhh, ws, src, dst, N, E, stream);
  // layers 2-4: GRU-gated, ping-pong hA/hB
  run_layer(hA, hB, Wl[1], asl[1], adl[1], Hc[1], true,  bih, bhh, ws, src, dst, N, E, stream);
  run_layer(hB, hA, Wl[2], asl[2], adl[2], Hc[2], true,  bih, bhh, ws, src, dst, N, E, stream);
  run_layer(hA, hB, Wl[3], asl[3], adl[3], Hc[3], true,  bih, bhh, ws, src, dst, N, E, stream);

  out_head_k<<<(N + 255) / 256, 256, 0, stream>>>(hB, W5, b5, (float*)d_out, N);
  (void)n_in; (void)out_size; (void)ws_size;
}
